// LigerFusedGRPOLoss_56813827392019
// MI455X (gfx1250) — matrix operand order repaired and checked by reading, not compiled, not run on hardware
//
#include <hip/hip_runtime.h>
#include <hip/hip_bf16.h>

// Problem constants (B,T,H,V) = (4,512,2048,64000)
#define BB   4
#define TT   512
#define HH   2048
#define VV   64000
#define MM   (BB*TT)          // 2048 token rows
#define BM   128              // M tile
#define BN   128              // N (vocab) tile
#define KC   64               // K slice staged in LDS
#define KP   72               // padded LDS row (bf16 elems); 72*2=144B = 9*16B -> keeps 16B align
#define KSTEPS (HH/KC)        // 32
#define NCHUNK (VV/BN)        // 500
#define MTILES (MM/BM)        // 16
#define BETA_C   0.04f
#define EPS_LO_C 0.2f
#define EPS_HI_C 0.2f

typedef __bf16 v16bf __attribute__((ext_vector_type(16)));
typedef __bf16 v2bf  __attribute__((ext_vector_type(2)));
typedef float  v8f   __attribute__((ext_vector_type(8)));
typedef unsigned int uint;

__device__ __forceinline__ unsigned short f2bf(float f) {
    unsigned u = __float_as_uint(f);
    u += 0x7FFFu + ((u >> 16) & 1u);      // round-to-nearest-even
    return (unsigned short)(u >> 16);
}

__device__ __forceinline__ uint pk2bf(float lo, float hi) {
#if __has_builtin(__builtin_amdgcn_cvt_pk_bf16_f32)
    union { v2bf v; uint u; } c;
    c.v = __builtin_amdgcn_cvt_pk_bf16_f32(lo, hi);
    return c.u;
#else
    return (uint)f2bf(lo) | ((uint)f2bf(hi) << 16);
#endif
}

// ---------------------------------------------------------------------------
// Kernel 1: fused bf16-WMMA GEMM + online row (max, sum-exp) partials.
// Double-buffered LDS software pipeline: global loads for slice k+1 are in
// flight while slice k's 16 WMMAs (per wave) execute; cvt+ds_store into the other buffer
// happens after the WMMAs; one workgroup barrier per K slice.
// grid = (MTILES, NCHUNK), block = 256 (8 waves)
// pmax/psum layout: [chunk][token]  (coalesced writes & reads)
// ---------------------------------------------------------------------------
__global__ __launch_bounds__(256)
void grpo_gemm_lse_kernel(const float* __restrict__ hs,
                          const float* __restrict__ w,
                          float* __restrict__ pmax,
                          float* __restrict__ psum) {
    __shared__ unsigned short As[2][BM * KP];   // 2 x 18 KB
    __shared__ unsigned short Bs[2][BN * KP];   // 2 x 18 KB
    __shared__ float pm[2][BM];
    __shared__ float ps[2][BM];

    const int tid  = threadIdx.x;
    const int lane = tid & 31;
    const int wid  = tid >> 5;
    const int mw   = (wid & 3) * 32;   // wave M offset inside tile
    const int nwi  = wid >> 2;         // 0..1
    const int nw   = nwi * 64;         // wave N offset inside tile

    const int m0 = blockIdx.x * BM;
    const int n0 = blockIdx.y * BN;

    v8f acc[2][4];
    const v8f vzero = {0.f,0.f,0.f,0.f,0.f,0.f,0.f,0.f};
    #pragma unroll
    for (int i = 0; i < 2; ++i)
        #pragma unroll
        for (int j = 0; j < 4; ++j) acc[i][j] = vzero;

    float4 ra[8], rb[8];

    auto load_chunk = [&](int kc) {
        #pragma unroll
        for (int i = 0; i < 8; ++i) {
            int idx = i * 256 + tid;           // 0..2047 float4 slots
            int row = idx >> 4;                // 0..127
            int c4  = idx & 15;                // float4 column
            ra[i] = *reinterpret_cast<const float4*>(
                hs + (size_t)(m0 + row) * HH + kc * KC + c4 * 4);
            const float* pb = w + (size_t)(n0 + row) * HH + kc * KC + c4 * 4;
            rb[i] = *reinterpret_cast<const float4*>(pb);
            if (kc + 1 < KSTEPS) __builtin_prefetch(pb + KC, 0, 0);
        }
    };

    auto store_chunk = [&](int buf) {
        #pragma unroll
        for (int i = 0; i < 8; ++i) {
            int idx = i * 256 + tid;
            int row = idx >> 4;
            int c4  = idx & 15;
            uint2 pa, pb;
            pa.x = pk2bf(ra[i].x, ra[i].y);
            pa.y = pk2bf(ra[i].z, ra[i].w);
            pb.x = pk2bf(rb[i].x, rb[i].y);
            pb.y = pk2bf(rb[i].z, rb[i].w);
            *reinterpret_cast<uint2*>(&As[buf][row * KP + c4 * 4]) = pa;
            *reinterpret_cast<uint2*>(&Bs[buf][row * KP + c4 * 4]) = pb;
        }
    };

    // pipeline prologue
    load_chunk(0);
    store_chunk(0);
    __syncthreads();

    const int klo = (lane & 16) ? 8 : 0;          // A lane K base (16-bit A layout)
    const int khi = (lane & 16) ? 8 : 0;          // B lane K base (same layout as A)
    const int lr  = lane & 15;

    for (int kc = 0; kc < KSTEPS; ++kc) {
        const int cur  = kc & 1;
        const bool more = (kc + 1 < KSTEPS);
        if (more) load_chunk(kc + 1);       // global loads in flight during WMMAs

        #pragma unroll
        for (int ks = 0; ks < KC; ks += 32) {
            union { uint4 u[2]; v16bf v; } ua[2], ub[4];
            #pragma unroll
            for (int mt = 0; mt < 2; ++mt) {
                const int r = mw + mt * 16 + lr;
                ua[mt].u[0] = *reinterpret_cast<const uint4*>(&As[cur][r * KP + ks + klo]);
                ua[mt].u[1] = *reinterpret_cast<const uint4*>(&As[cur][r * KP + ks + klo + 16]);
            }
            #pragma unroll
            for (int nt = 0; nt < 4; ++nt) {
                const int n = nw + nt * 16 + lr;
                const uint4* q = reinterpret_cast<const uint4*>(&Bs[cur][n * KP + ks + khi]);
                ub[nt].u[0] = q[0];
                ub[nt].u[1] = q[2];
            }
            #pragma unroll
            for (int mt = 0; mt < 2; ++mt)
                #pragma unroll
                for (int nt = 0; nt < 4; ++nt)
                    acc[mt][nt] = __builtin_amdgcn_wmma_f32_16x16x32_bf16(
                        false, ua[mt].v, false, ub[nt].v,
                        (short)0, acc[mt][nt], false, false);
        }

        if (more) store_chunk(1 - cur);     // fill the other buffer
        __syncthreads();
    }

    // ---- in-register row (max, sum-exp) over this 128-wide vocab chunk ----
    #pragma unroll
    for (int mt = 0; mt < 2; ++mt) {
        v8f rm;
        #pragma unroll
        for (int c = 0; c < 8; ++c) {
            float v = acc[mt][0][c];
            v = fmaxf(v, acc[mt][1][c]);
            v = fmaxf(v, acc[mt][2][c]);
            v = fmaxf(v, acc[mt][3][c]);
            // butterfly max within 16-lane group (rows differ across halves)
            v = fmaxf(v, __shfl_xor(v, 1, 32));
            v = fmaxf(v, __shfl_xor(v, 2, 32));
            v = fmaxf(v, __shfl_xor(v, 4, 32));
            v = fmaxf(v, __shfl_xor(v, 8, 32));
            rm[c] = v;
        }
        v8f se;
        #pragma unroll
        for (int c = 0; c < 8; ++c) {
            float s = __expf(acc[mt][0][c] - rm[c])
                    + __expf(acc[mt][1][c] - rm[c])
                    + __expf(acc[mt][2][c] - rm[c])
                    + __expf(acc[mt][3][c] - rm[c]);
            s += __shfl_xor(s, 1, 32);
            s += __shfl_xor(s, 2, 32);
            s += __shfl_xor(s, 4, 32);
            s += __shfl_xor(s, 8, 32);
            se[c] = s;
        }
        if ((lane & 15) == 0) {
            const int rbase = mw + mt * 16 + ((lane >> 4) << 3);
            #pragma unroll
            for (int c = 0; c < 8; ++c) {
                pm[nwi][rbase + c] = rm[c];
                ps[nwi][rbase + c] = se[c];
            }
        }
    }
    __syncthreads();

    // ---- merge the two N halves, write chunk partials ----
    if (tid < BM) {
        float ma = pm[0][tid], mb = pm[1][tid];
        float sa = ps[0][tid], sb = ps[1][tid];
        float M = fmaxf(ma, mb);
        float S = sa * __expf(ma - M) + sb * __expf(mb - M);
        const int gtok = m0 + tid;
        pmax[(size_t)blockIdx.y * MM + gtok] = M;
        psum[(size_t)blockIdx.y * MM + gtok] = S;
    }
}

// ---------------------------------------------------------------------------
// Kernel 2: selected-token logit, fp32 dot, one wave per token.
// ---------------------------------------------------------------------------
__global__ __launch_bounds__(256)
void grpo_sel_logit_kernel(const float* __restrict__ hs,
                           const float* __restrict__ w,
                           const int* __restrict__ ids,
                           float* __restrict__ sel) {
    const int tid  = threadIdx.x;
    const int lane = tid & 31;
    const int tok  = (blockIdx.x * blockDim.x + tid) >> 5;
    if (tok >= MM) return;
    const float* hp = hs + (size_t)tok * HH;
    const float* wp = w  + (size_t)ids[tok] * HH;
    float a0 = 0.f, a1 = 0.f, a2 = 0.f, a3 = 0.f;
    for (int k = lane * 4; k < HH; k += 32 * 4) {
        float4 x = *reinterpret_cast<const float4*>(hp + k);
        float4 y = *reinterpret_cast<const float4*>(wp + k);
        a0 = fmaf(x.x, y.x, a0);
        a1 = fmaf(x.y, y.y, a1);
        a2 = fmaf(x.z, y.z, a2);
        a3 = fmaf(x.w, y.w, a3);
    }
    float s = (a0 + a1) + (a2 + a3);
    s += __shfl_xor(s, 1, 32);
    s += __shfl_xor(s, 2, 32);
    s += __shfl_xor(s, 4, 32);
    s += __shfl_xor(s, 8, 32);
    s += __shfl_xor(s, 16, 32);
    if (lane == 0) sel[tok] = s;
}

// ---------------------------------------------------------------------------
// Kernel 3: merge chunk partials -> logsumexp -> per-token GRPO loss (masked)
// ---------------------------------------------------------------------------
__global__ __launch_bounds__(256)
void grpo_token_loss_kernel(const float* __restrict__ pmax,
                            const float* __restrict__ psum,
                            const float* __restrict__ sel,
                            const float* __restrict__ amask,
                            const float* __restrict__ adv,
                            const float* __restrict__ oldlp,
                            const float* __restrict__ reflp,
                            float* __restrict__ lossm) {
    const int bt = blockIdx.x * blockDim.x + threadIdx.x;
    if (bt >= MM) return;
    float gm = pmax[bt];
    float gs = psum[bt];
    for (int c = 1; c < NCHUNK; ++c) {
        float m = pmax[(size_t)c * MM + bt];
        float s = psum[(size_t)c * MM + bt];
        if (m > gm) { gs = gs * __expf(gm - m) + s; gm = m; }
        else        { gs += s * __expf(m - gm); }
    }
    const float lse  = gm + __logf(gs);
    const float logp = sel[bt] - lse;
    const float c1 = __expf(logp - oldlp[bt]);
    const float c2 = fminf(fmaxf(c1, 1.0f - EPS_LO_C), 1.0f + EPS_HI_C);
    const float a  = adv[bt >> 9];               // bt / T
    float pl = -fminf(c1 * a, c2 * a);
    const float d = reflp[bt] - logp;
    pl += BETA_C * (__expf(d) - d - 1.0f);
    lossm[bt] = pl * amask[bt];
}

// ---------------------------------------------------------------------------
// Kernel 4: masked per-sequence mean, then mean over batch -> scalar
// ---------------------------------------------------------------------------
__global__ __launch_bounds__(256)
void grpo_reduce_kernel(const float* __restrict__ lossm,
                        const float* __restrict__ amask,
                        float* __restrict__ out) {
    __shared__ float sl[256];
    __shared__ float sm[256];
    const int tid = threadIdx.x;
    float tot = 0.f;
    for (int b = 0; b < BB; ++b) {
        float ls = 0.f, ms = 0.f;
        for (int t = tid; t < TT; t += 256) {
            ls += lossm[b * TT + t];
            ms += amask[b * TT + t];
        }
        sl[tid] = ls; sm[tid] = ms;
        __syncthreads();
        for (int off = 128; off > 0; off >>= 1) {
            if (tid < off) { sl[tid] += sl[tid + off]; sm[tid] += sm[tid + off]; }
            __syncthreads();
        }
        if (tid == 0) tot += sl[0] / fmaxf(sm[0], 1.0f);
        __syncthreads();
    }
    if (tid == 0) out[0] = tot * (1.0f / BB);
}

// ---------------------------------------------------------------------------
extern "C" void kernel_launch(void* const* d_in, const int* in_sizes, int n_in,
                              void* d_out, int out_size, void* d_ws, size_t ws_size,
                              hipStream_t stream) {
    const float* hs    = (const float*)d_in[0];   // (B,T,H)
    const float* w     = (const float*)d_in[1];   // (V,H)
    const float* amask = (const float*)d_in[2];   // (B,T)
    const float* adv   = (const float*)d_in[3];   // (B,)
    const float* oldlp = (const float*)d_in[4];   // (B,T)
    const float* reflp = (const float*)d_in[5];   // (B,T)
    const int*   ids   = (const int*)d_in[6];     // (B,T)

    float* pmax  = (float*)d_ws;                       // [NCHUNK][MM]
    float* psum  = pmax + (size_t)NCHUNK * MM;         // [NCHUNK][MM]
    float* sel   = psum + (size_t)NCHUNK * MM;         // [MM]
    float* lossm = sel + MM;                           // [MM]
    float* out   = (float*)d_out;

    dim3 g1(MTILES, NCHUNK);
    grpo_gemm_lse_kernel<<<g1, 256, 0, stream>>>(hs, w, pmax, psum);

    grpo_sel_logit_kernel<<<(MM * 32) / 256, 256, 0, stream>>>(hs, w, ids, sel);

    grpo_token_loss_kernel<<<MM / 256, 256, 0, stream>>>(pmax, psum, sel, amask,
                                                         adv, oldlp, reflp, lossm);

    grpo_reduce_kernel<<<1, 256, 0, stream>>>(lossm, amask, out);
}
